// VoteQN2_39659728011948
// MI455X (gfx1250) — matrix operand order repaired and checked by reading, not compiled, not run on hardware
//
#include <hip/hip_runtime.h>
#include <hip/hip_bf16.h>
#include <math.h>

// ---------------------------------------------------------------------------
// CDNA5 / gfx1250 wave32 helpers
// ---------------------------------------------------------------------------
typedef __attribute__((ext_vector_type(16))) __bf16 v16bf;
typedef __attribute__((ext_vector_type(8)))  __bf16 v8bf;
typedef __attribute__((ext_vector_type(4)))  __bf16 v4bf;
typedef __attribute__((ext_vector_type(8)))  float  v8f;
typedef __attribute__((ext_vector_type(4)))  float  v4f;

__device__ inline unsigned ballot32(bool p) {
#if __has_builtin(__builtin_amdgcn_ballot_w32)
    return __builtin_amdgcn_ballot_w32(p);
#else
    return (unsigned)__ballot(p ? 1 : 0);
#endif
}
__device__ inline void wave_bar() {
#if __has_builtin(__builtin_amdgcn_wave_barrier)
    __builtin_amdgcn_wave_barrier();
#endif
}
// SWAPX16 ds_swizzle: exchange value with lane^16 (group-of-32 xor pattern).
__device__ inline float swap_half(float x) {
#if __has_builtin(__builtin_amdgcn_ds_swizzle)
    return __int_as_float(__builtin_amdgcn_ds_swizzle(__float_as_int(x), 0x401f));
#else
    return __shfl_xor(x, 16, 32);
#endif
}

#define EPSV 1e-5f

// ---------------------------------------------------------------------------
// Farthest point sampling: one block per batch, running-min dists in LDS.
// ---------------------------------------------------------------------------
__global__ __launch_bounds__(256)
void fps_kernel(const float* __restrict__ xyz, int ld, int N, int npoint,
                int* __restrict__ fps_idx, float* __restrict__ new_xyz) {
    __shared__ float dists[2048];
    __shared__ float rv[256];
    __shared__ int   ri[256];
    __shared__ float fpt[3];
    __shared__ int   farIdx;

    const int b   = blockIdx.x;
    const int tid = threadIdx.x;
    const float* P = xyz + (size_t)b * N * ld;

    for (int i = tid; i < N; i += 256) dists[i] = 1e10f;
    if (tid == 0) farIdx = 0;
    __syncthreads();

    for (int it = 0; it < npoint; ++it) {
        const int far = farIdx;
        if (tid == 0) {
            fps_idx[b * npoint + it] = far;
            const float* fp = P + (size_t)far * ld;
            fpt[0] = fp[0]; fpt[1] = fp[1]; fpt[2] = fp[2];
            float* nx = new_xyz + (size_t)(b * npoint + it) * 3;
            nx[0] = fp[0]; nx[1] = fp[1]; nx[2] = fp[2];
        }
        __syncthreads();
        const float fx = fpt[0], fy = fpt[1], fz = fpt[2];
        float best = -1.0f; int bi = 0;
        for (int i = tid; i < N; i += 256) {
            const float* q = P + (size_t)i * ld;
            float dx = q[0] - fx, dy = q[1] - fy, dz = q[2] - fz;
            float d  = dx * dx + dy * dy + dz * dz;
            float nd = fminf(dists[i], d);
            dists[i] = nd;
            if (nd > best) { best = nd; bi = i; }
        }
        rv[tid] = best; ri[tid] = bi;
        __syncthreads();
        for (int s = 128; s > 0; s >>= 1) {
            if (tid < s && rv[tid + s] > rv[tid]) { rv[tid] = rv[tid + s]; ri[tid] = ri[tid + s]; }
            __syncthreads();
        }
        if (tid == 0) farIdx = ri[0];
        __syncthreads();
    }
}

// ---------------------------------------------------------------------------
// Ball query + gather + concat, one wave32 per (batch, center).
// Writes rows of length Kpad (>= 3+Cfeat, multiple of 32), zero padded.
// ---------------------------------------------------------------------------
__global__ __launch_bounds__(256)
void bq_build_kernel(const float* __restrict__ xyz, int ldx,
                     const float* __restrict__ feats, int ldf, int Cfeat,
                     const float* __restrict__ new_xyz,
                     int B, int N, int S, float radius, int Kpad,
                     float* __restrict__ x0) {
    __shared__ int sidx[8][16];
    const int wib  = threadIdx.x >> 5;
    const int w    = blockIdx.x * 8 + wib;
    if (w >= B * S) return;               // wave-uniform exit
    const int lane = threadIdx.x & 31;
    const int b    = w / S;
    int* idx = sidx[wib];

    const float* P  = xyz + (size_t)b * N * ldx;
    const float cx = new_xyz[(size_t)w * 3 + 0];
    const float cy = new_xyz[(size_t)w * 3 + 1];
    const float cz = new_xyz[(size_t)w * 3 + 2];
    const float r2 = radius * radius;

    int cnt = 0;
    for (int base = 0; base < N && cnt < 16; base += 32) {
        const int j = base + lane;
        bool ok = false;
        if (j < N) {
            const float* q = P + (size_t)j * ldx;
            float dx = q[0] - cx, dy = q[1] - cy, dz = q[2] - cz;
            ok = (dx * dx + dy * dy + dz * dz) < r2;
        }
        const unsigned m  = ballot32(ok);
        const int pos = cnt + __popc(m & ((1u << lane) - 1u));
        if (ok && pos < 16) idx[pos] = j;
        cnt += __popc(m);
    }
    if (cnt > 16) cnt = 16;
    if (cnt == 0) { if (lane == 0) idx[0] = 0; cnt = 1; }
    wave_bar();
    const int i0 = idx[0];
    if (lane >= cnt && lane < 16) idx[lane] = i0;   // pad with first hit
    wave_bar();

    const float invr    = 1.0f / radius;
    const size_t rowbase = (size_t)w * 16;
    for (int k = 0; k < 16; ++k) {
        const int j = idx[k];
        const float* Prow = P + (size_t)j * ldx;
        const float* Frow = feats + ((size_t)b * N + j) * ldf;
        float* orow = x0 + (rowbase + k) * Kpad;
        if (lane == 0) {
            orow[0] = (Prow[0] - cx) * invr;
            orow[1] = (Prow[1] - cy) * invr;
            orow[2] = (Prow[2] - cz) * invr;
        }
        for (int c = lane; c < Kpad - 3; c += 32)
            orow[3 + c] = (c < Cfeat) ? Frow[c] : 0.0f;
    }
}

// ---------------------------------------------------------------------------
// Pre-swizzle fp32 weights (Cin x Cout) into per-lane bf16 B fragments,
// zero padded to Kpad rows. Layout: [ct][kt][lane][16 bf16] so a lane's
// fragment is 32 contiguous bytes (two b128 loads, fully coalesced).
// ---------------------------------------------------------------------------
__global__ void pack_weights_kernel(const float* __restrict__ W, int Cin, int Cout,
                                    int Kpad, __bf16* __restrict__ Wpack) {
    const int KT    = Kpad >> 5;
    const int total = (Cout >> 4) * KT * 32;
    const int t = blockIdx.x * blockDim.x + threadIdx.x;
    if (t >= total) return;
    const int lane = t & 31;
    const int kt   = (t >> 5) % KT;
    const int ct   = (t >> 5) / KT;
    const int n    = ct * 16 + (lane & 15);
    __bf16* dst = Wpack + (size_t)t * 16;
#pragma unroll
    for (int e = 0; e < 16; ++e) {
        const int v = e >> 1, h = e & 1;
        const int k = kt * 32 + ((lane < 16) ? 0 : 8) + e + ((e >= 8) ? 8 : 0);
        dst[e] = (__bf16)((k < Cin) ? W[(size_t)k * Cout + n] : 0.0f);
    }
}

// ---------------------------------------------------------------------------
// GEMM: one wave per 16x64 output strip, 4 bf16 WMMAs per 32-wide K step.
// A tile staged through LDS (b128 loads + fused BN/ReLU + bf16 cvt), read
// back as two ds_load_b128 fragments. B from pre-swizzled Wpack. Per-channel
// sum/sumsq accumulated with a ds_swizzle half-wave reduction + atomics.
// HAS_AFFINE selects the fused prev-layer BN+ReLU path at compile time so
// both variants have branch-free inner loops.
// ---------------------------------------------------------------------------
template <bool HAS_AFFINE>
__global__ __launch_bounds__(256)
void mlp_wmma_kernel(const float* __restrict__ X, int ldX, int Kpad,
                     const __bf16* __restrict__ Wpack,
                     const float* __restrict__ inScale, const float* __restrict__ inBias,
                     int R, int Cout,
                     float* __restrict__ Y,
                     float* __restrict__ statSum, float* __restrict__ statSq) {
    __shared__ __align__(16) __bf16 tiles[8][16 * 32];
    const int colGroups = Cout >> 6;                 // 64 columns per wave
    const int total     = (R >> 4) * colGroups;
    const int wib       = threadIdx.x >> 5;
    const int w         = blockIdx.x * 8 + wib;
    if (w >= total) return;                          // wave-uniform

    const int rowTile = w / colGroups;
    const int cg      = w - rowTile * colGroups;
    const int lane    = threadIdx.x & 31;
    const bool lo     = lane < 16;
    const int  mn     = lane & 15;
    const int rowBase = rowTile << 4;
    const int colBase = cg << 6;
    const int KT      = Kpad >> 5;
    __bf16* tile = tiles[wib];

    v8f acc0 = {}, acc1 = {}, acc2 = {}, acc3 = {};

    for (int kt = 0; kt < KT; ++kt) {
        const int k0 = kt << 5;
        // -- cooperative A-tile load: 4 x float4 per lane, coalesced --
#pragma unroll
        for (int g = 0; g < 4; ++g) {
            const int flat = g * 128 + lane * 4;
            const int row  = flat >> 5;
            const int col  = flat & 31;
            v4f xv = *(const v4f*)(X + (size_t)(rowBase + row) * ldX + k0 + col);
            if constexpr (HAS_AFFINE) {               // fused BN + ReLU of prev layer
                v4f sc = *(const v4f*)(inScale + k0 + col);
                v4f bi = *(const v4f*)(inBias + k0 + col);
#pragma unroll
                for (int q = 0; q < 4; ++q) xv[q] = fmaxf(xv[q] * sc[q] + bi[q], 0.0f);
            }
            v4bf cv = { (__bf16)xv[0], (__bf16)xv[1], (__bf16)xv[2], (__bf16)xv[3] };
            *(v4bf*)&tile[row * 32 + col] = cv;
        }
        wave_bar();
        // -- A fragment: two 8x bf16 runs per the 16x32 bf16 layout --
        const int off = lo ? 0 : 8;
        v8bf a0 = *(const v8bf*)&tile[mn * 32 + off];
        v8bf a1 = *(const v8bf*)&tile[mn * 32 + off + 16];
        v16bf a = __builtin_shufflevector(a0, a1, 0, 1, 2, 3, 4, 5, 6, 7,
                                                 8, 9, 10, 11, 12, 13, 14, 15);
        // -- 4 column tiles: B fragments are 32 contiguous bytes per lane --
        const __bf16* wp0 = Wpack + (((size_t)(cg * 4) * KT + kt) * 32 + lane) * 16;
#pragma unroll
        for (int j = 0; j < 4; ++j) {
            const __bf16* wp = wp0 + (size_t)j * KT * 512;
            v8bf b0 = *(const v8bf*)wp;
            v8bf b1 = *(const v8bf*)(wp + 8);
            v16bf bb = __builtin_shufflevector(b0, b1, 0, 1, 2, 3, 4, 5, 6, 7,
                                                      8, 9, 10, 11, 12, 13, 14, 15);
            v8f* accp = (j == 0) ? &acc0 : (j == 1) ? &acc1 : (j == 2) ? &acc2 : &acc3;
            *accp = __builtin_amdgcn_wmma_f32_16x16x32_bf16(
                false, a, false, bb, (short)0, *accp, false, false);
        }
        wave_bar();                                   // tile reused next step
    }

    // -- store D tiles + per-channel stats --
#pragma unroll
    for (int j = 0; j < 4; ++j) {
        const v8f accv = (j == 0) ? acc0 : (j == 1) ? acc1 : (j == 2) ? acc2 : acc3;
        const int cb = colBase + j * 16;
        float ps = 0.0f, pq = 0.0f;
#pragma unroll
        for (int r = 0; r < 8; ++r) {
            const int m2 = lo ? r : (r + 8);
            const float vv = accv[r];
            Y[(size_t)(rowBase + m2) * Cout + cb + mn] = vv;
            ps += vv; pq += vv * vv;
        }
        ps += swap_half(ps);                          // combine lane pairs (n, n+16)
        pq += swap_half(pq);
        if (lo) {
            atomicAdd(&statSum[cb + mn], ps);
            atomicAdd(&statSq[cb + mn], pq);
        }
    }
}

// ---------------------------------------------------------------------------
// (sum, sumsq, gamma, beta) -> per-channel affine (scale, bias).
// ---------------------------------------------------------------------------
__global__ void bn_finalize_kernel(const float* __restrict__ sum, const float* __restrict__ sq,
                                   const float* __restrict__ gamma, const float* __restrict__ beta,
                                   int C, float invR,
                                   float* __restrict__ scale, float* __restrict__ bias) {
    const int c = blockIdx.x * blockDim.x + threadIdx.x;
    if (c < C) {
        const float mu  = sum[c] * invR;
        const float var = sq[c] * invR - mu * mu;
        const float s   = gamma[c] * rsqrtf(var + EPSV);
        scale[c] = s;
        bias[c]  = beta[c] - mu * s;
    }
}

// ---------------------------------------------------------------------------
// y = max_k relu(Y*scale+bias), Y is (BS,16,C).
// ---------------------------------------------------------------------------
__global__ void norm_relu_max_kernel(const float* __restrict__ Y,
                                     const float* __restrict__ scale, const float* __restrict__ bias,
                                     int BS, int C, float* __restrict__ feats) {
    const int t = blockIdx.x * blockDim.x + threadIdx.x;
    if (t >= BS * C) return;
    const int row = t / C, c = t - row * C;
    const float s = scale[c], bb = bias[c];
    const float* p = Y + ((size_t)row * 16) * C + c;
    float m = 0.0f;                                   // relu outputs are >= 0
#pragma unroll
    for (int k = 0; k < 16; ++k) m = fmaxf(m, fmaxf(p[(size_t)k * C] * s + bb, 0.0f));
    feats[t] = m;
}

__global__ void final_out_kernel(const float* __restrict__ f, const float* __restrict__ mask,
                                 float* __restrict__ out, int n) {
    const int t = blockIdx.x * blockDim.x + threadIdx.x;
    if (t < n) out[t] = f[t] * mask[t];
}

// ---------------------------------------------------------------------------
// Host orchestration
// ---------------------------------------------------------------------------
extern "C" void kernel_launch(void* const* d_in, const int* in_sizes, int n_in,
                              void* d_out, int out_size, void* d_ws, size_t ws_size,
                              hipStream_t stream) {
    (void)in_sizes; (void)n_in; (void)out_size; (void)ws_size;
    const float* obs  = (const float*)d_in[0];
    const float* mask = (const float*)d_in[1];

    const int B = 64;
    const int   npoints[3] = {64, 16, 1};
    const float radii[3]   = {1.2f, 3.6f, 3.6f};
    const int   Ns[3]      = {2048, 64, 16};
    const int   dims[3][4] = {{259, 64, 64, 128}, {131, 128, 128, 256}, {259, 128, 64, 64}};
    const int   Kpad0[3]   = {288, 160, 288};          // layer-0 K padded to 32

    // --- carve workspace (floats, all chunks 16B aligned) ---
    float* wsf = (float*)d_ws;
    size_t off = 0;
    auto carve = [&](size_t nfloats) { float* p = wsf + off; off += nfloats; return p; };
    float*  x0      = carve((size_t)64 * 64 * 16 * 288);  // padded gather buffer (max)
    float*  Ya      = carve((size_t)64 * 64 * 16 * 128);
    float*  Yb      = carve((size_t)64 * 64 * 16 * 128);
    float*  nx[3]   = { carve((size_t)64 * 64 * 3 + 4), carve((size_t)64 * 16 * 3 + 4), carve(256) };
    float*  fo[3]   = { carve((size_t)64 * 64 * 128), carve((size_t)64 * 16 * 256), carve((size_t)64 * 64) };
    float*  statSum = carve(256);
    float*  statSq  = carve(256);     // contiguous after statSum -> single memset
    float*  bnScale = carve(256);
    float*  bnBias  = carve(256);
    __bf16* Wpack   = (__bf16*)carve(32768);              // 64K bf16 elements max
    int*    fpsIdx  = (int*)carve(4096);

    for (int s = 0; s < 3; ++s) {
        const int   S  = npoints[s];
        const int   N  = Ns[s];
        const float rr = radii[s];
        const float* xyz;  int ldx;
        const float* fin;  int ldf; int Cfeat;
        if (s == 0) { xyz = obs; ldx = 259; fin = obs + 3; ldf = 259; Cfeat = 256; }
        else        { xyz = nx[s - 1]; ldx = 3; fin = fo[s - 1]; ldf = dims[s - 1][3]; Cfeat = dims[s - 1][3]; }

        fps_kernel<<<B, 256, 0, stream>>>(xyz, ldx, N, S, fpsIdx, nx[s]);

        const int centers = B * S;
        bq_build_kernel<<<(centers + 7) / 8, 256, 0, stream>>>(
            xyz, ldx, fin, ldf, Cfeat, nx[s], B, N, S, rr, Kpad0[s], x0);

        const int R = B * S * 16;
        const float* Xl = x0;
        int ldX  = Kpad0[s];
        int Kpad = Kpad0[s];
        float* Ys[2] = {Ya, Yb};
        for (int l = 0; l < 3; ++l) {
            const int Cin  = dims[s][l];
            const int Cout = dims[s][l + 1];
            const float* Wp = (const float*)d_in[2 + s * 9 + l * 3 + 0];
            const float* gp = (const float*)d_in[2 + s * 9 + l * 3 + 1];
            const float* bp = (const float*)d_in[2 + s * 9 + l * 3 + 2];

            const int packThreads = (Cout / 16) * (Kpad / 32) * 32;
            pack_weights_kernel<<<(packThreads + 255) / 256, 256, 0, stream>>>(
                Wp, Cin, Cout, Kpad, Wpack);

            hipMemsetAsync(statSum, 0, 512 * sizeof(float), stream);   // sum + sq

            const int totalWaves = (R / 16) * (Cout / 64);
            const int blocks     = (totalWaves + 7) / 8;
            if (l == 0) {
                mlp_wmma_kernel<false><<<blocks, 256, 0, stream>>>(
                    Xl, ldX, Kpad, Wpack, nullptr, nullptr,
                    R, Cout, Ys[l & 1], statSum, statSq);
            } else {
                mlp_wmma_kernel<true><<<blocks, 256, 0, stream>>>(
                    Xl, ldX, Kpad, Wpack, bnScale, bnBias,
                    R, Cout, Ys[l & 1], statSum, statSq);
            }

            bn_finalize_kernel<<<(Cout + 63) / 64, 64, 0, stream>>>(
                statSum, statSq, gp, bp, Cout, 1.0f / (float)R, bnScale, bnBias);

            Xl   = Ys[l & 1];
            ldX  = Cout;
            Kpad = Cout;
        }
        const int Clast = dims[s][3];
        norm_relu_max_kernel<<<(B * S * Clast + 255) / 256, 256, 0, stream>>>(
            Xl, bnScale, bnBias, B * S, Clast, fo[s]);
    }

    final_out_kernel<<<(4096 + 255) / 256, 256, 0, stream>>>(fo[2], mask, (float*)d_out, 4096);
}
